// Head_16226386444373
// MI455X (gfx1250) — matrix operand order repaired and checked by reading, not compiled, not run on hardware
//
#include <hip/hip_runtime.h>

// Problem constants (match reference)
#define BB 8
#define TT 4096
#define HH 64
#define EE 16
#define LDS_STRIDE 40   // padded row stride (halves): 80B = 20 banks, kills
                        // the 2-way store conflict a 64B stride would have

typedef __attribute__((ext_vector_type(16))) _Float16 v16h;
typedef __attribute__((ext_vector_type(8)))  _Float16 v8h;
typedef __attribute__((ext_vector_type(8)))  float    v8f;

__device__ __forceinline__ v16h make16(v8h lo, v8h hi) {
  return __builtin_shufflevector(lo, hi, 0,1,2,3,4,5,6,7,8,9,10,11,12,13,14,15);
}

// A-matrix 16x32 f16 fragment (ISA 7.12.2): lanes 0-15 hold K={0..7,16..23},
// lanes 16-31 hold K={8..15,24..31}. Per lane: two contiguous 16B loads.
__device__ __forceinline__ v16h load_afrag(const _Float16* row, int lhalf) {
  const int kb = lhalf * 8;
  v8h lo = *(const v8h*)(row + kb);
  v8h hi = *(const v8h*)(row + kb + 16);
  return make16(lo, hi);
}

// B-matrix 32x16 f16 fragment: lanes 0-15 hold K={0..7,16..23},
// lanes 16-31 hold K={8..15,24..31} (matches A). Two 16B loads per lane.
__device__ __forceinline__ v16h load_bfrag(const _Float16* row, int lhalf) {
  const int kb = lhalf * 8;
  v8h lo = *(const v8h*)(row + kb);
  v8h hi = *(const v8h*)(row + kb + 16);
  return make16(lo, hi);
}

#define WMMA_F16(a, b, c) \
  __builtin_amdgcn_wmma_f32_16x16x32_f16(false, (a), false, (b), (short)0, (c), false, false)

// scores^(-1/16) then exp (softmax numerator). Scores strictly positive.
// Hardware transcendentals: v_log_f32 (log2) and v_exp_f32 (exp2).
__device__ __forceinline__ float pow_exp(float s) {
  float p = __builtin_amdgcn_exp2f(-0.0625f * __builtin_amdgcn_logf(s)); // s^(-1/16)
  return __builtin_amdgcn_exp2f(p * 1.4426950408889634f);                // exp(p)
}

// Wave-local LDS fence: DS ops from one wave are processed in order, so
// cross-lane RAW within the wave only needs DScnt==0 plus a compiler
// scheduling fence. Avoids block-wide s_barrier convoying.
__device__ __forceinline__ void wave_lds_fence_raw() {
  __builtin_amdgcn_wave_barrier();
  asm volatile("s_wait_dscnt 0x0" ::: "memory");
}
__device__ __forceinline__ void wave_lds_fence_war() {
  // WAR: per-wave LDS ops stay in order; only stop compiler reordering.
  __builtin_amdgcn_wave_barrier();
}

// ---------------------------------------------------------------------------
// Kernel 1: q = x@Wq, k = x@Wk (v == k per reference). Emit f16 row-major
// qh/kh plus h-major vT so every WMMA operand later is a contiguous load.
// ---------------------------------------------------------------------------
__global__ __launch_bounds__(256) void proj_kernel(
    const float* __restrict__ x, const float* __restrict__ Wq,
    const float* __restrict__ Wk, _Float16* __restrict__ qh,
    _Float16* __restrict__ kh, _Float16* __restrict__ vT) {
  __shared__ float sWq[EE * HH];
  __shared__ float sWk[EE * HH];
  for (int i = threadIdx.x; i < EE * HH; i += 256) {
    sWq[i] = Wq[i];
    sWk[i] = Wk[i];
  }
  __syncthreads();

  const int idx = blockIdx.x * 256 + threadIdx.x;   // over B*T
  const int b = idx >> 12;
  const int t = idx & (TT - 1);

  float xv[EE];
#pragma unroll
  for (int e = 0; e < EE; ++e) xv[e] = x[(size_t)idx * EE + e];

#pragma unroll 4
  for (int h = 0; h < HH; ++h) {
    float q = 0.f, k = 0.f;
#pragma unroll
    for (int e = 0; e < EE; ++e) {
      q += xv[e] * sWq[e * HH + h];
      k += xv[e] * sWk[e * HH + h];
    }
    qh[(size_t)idx * HH + h] = (_Float16)q;
    kh[(size_t)idx * HH + h] = (_Float16)k;
    vT[((size_t)b * HH + h) * TT + t] = (_Float16)k;   // v = k, transposed
  }
}

// ---------------------------------------------------------------------------
// Kernel 2: Z[b,s] = sum_t exp( (q_t . k_s)^(-1/16) )   (softmax over axis=1)
// One wave per 16-wide s-tile; streams all 256 t-tiles with chained K=64 WMMA.
// ---------------------------------------------------------------------------
__global__ __launch_bounds__(256) void zsum_kernel(
    const _Float16* __restrict__ qh, const _Float16* __restrict__ kh,
    float* __restrict__ Z) {
  const int lane  = threadIdx.x & 31;
  const int wv    = threadIdx.x >> 5;
  const int gw    = blockIdx.x * 8 + wv;     // 8 waves / block
  const int b     = gw >> 8;                 // T/16 = 256 s-tiles per batch
  const int s0    = (gw & 255) << 4;
  const int lmod  = lane & 15;
  const int lhalf = lane >> 4;

  const _Float16* krow = kh + (size_t)(b * TT + s0 + lmod) * HH;
  const v16h bk0 = load_bfrag(krow, lhalf);        // h = 0..31
  const v16h bk1 = load_bfrag(krow + 32, lhalf);   // h = 32..63

  float zacc = 0.f;
  const _Float16* qbase = qh + (size_t)b * TT * HH;
  for (int t0 = 0; t0 < TT; t0 += 16) {
    const _Float16* qrow = qbase + (size_t)(t0 + lmod) * HH;
    v16h aq0 = load_afrag(qrow, lhalf);
    v16h aq1 = load_afrag(qrow + 32, lhalf);
    v8f d = {};
    d = WMMA_F16(aq0, bk0, d);
    d = WMMA_F16(aq1, bk1, d);
#pragma unroll
    for (int r = 0; r < 8; ++r) zacc += pow_exp(d[r]);
  }
  // D layout: lane l and l+16 cover the same column (M=r vs M=r+8).
  zacc += __shfl_xor(zacc, 16, 32);
  if (lhalf == 0) Z[b * TT + s0 + lmod] = zacc;
}

// ---------------------------------------------------------------------------
// Kernel 3: out[b,t,:] = sum_s exp(pow(q_t.k_s))/Z[b,s] * v[b,s,:]
// One wave per 16-row t-tile; streams s in chunks of 32. Score D-tiles are
// normalized, converted to f16 A-layout via a per-wave LDS round trip
// (wave-local fences only — no block barriers), then fed into 4 PV WMMAs
// (h tiles 0..3) with f32 accumulation held in registers.
// ---------------------------------------------------------------------------
__global__ __launch_bounds__(256) void attn_out_kernel(
    const _Float16* __restrict__ qh, const _Float16* __restrict__ kh,
    const _Float16* __restrict__ vT, const float* __restrict__ Z,
    float* __restrict__ out) {
  __shared__ _Float16 lds[8][16][LDS_STRIDE];   // per-wave 16(t) x 32(s) tile

  const int lane  = threadIdx.x & 31;
  const int wv    = threadIdx.x >> 5;
  const int gw    = blockIdx.x * 8 + wv;
  const int b     = gw >> 8;                 // T/16 = 256 t-tiles per batch
  const int t0    = (gw & 255) << 4;
  const int lmod  = lane & 15;
  const int lhalf = lane >> 4;

  const _Float16* qrow = qh + (size_t)(b * TT + t0 + lmod) * HH;
  const v16h aq0 = load_afrag(qrow, lhalf);
  const v16h aq1 = load_afrag(qrow + 32, lhalf);

  v8f acc[4] = {};   // h tiles 0,16,32,48

  for (int s0 = 0; s0 < TT; s0 += 32) {
#pragma unroll
    for (int sub = 0; sub < 2; ++sub) {
      const int ss = s0 + sub * 16;
      const _Float16* krow = kh + (size_t)(b * TT + ss + lmod) * HH;
      v16h bk0 = load_bfrag(krow, lhalf);
      v16h bk1 = load_bfrag(krow + 32, lhalf);
      v8f d = {};
      d = WMMA_F16(aq0, bk0, d);
      d = WMMA_F16(aq1, bk1, d);
      const float zinv = 1.0f / Z[b * TT + ss + lmod];
#pragma unroll
      for (int r = 0; r < 8; ++r) {
        float w = pow_exp(d[r]) * zinv;
        // D layout -> [t_local][s_local] in LDS
        lds[wv][r + lhalf * 8][sub * 16 + lmod] = (_Float16)w;
      }
    }
    wave_lds_fence_raw();   // make this wave's stores visible to its reads

    // Re-read weights in A-fragment layout (16x32, K = s_local)
    v16h aw;
    {
      const _Float16* lrow = &lds[wv][lmod][0];
      const int kb = lhalf * 8;
      v8h lo = *(const v8h*)(lrow + kb);
      v8h hi = *(const v8h*)(lrow + kb + 16);
      aw = make16(lo, hi);
    }
#pragma unroll
    for (int h4 = 0; h4 < 4; ++h4) {
      const _Float16* vrow = vT + (size_t)(b * HH + h4 * 16 + lmod) * TT + s0;
      v16h bv = load_bfrag(vrow, lhalf);   // contiguous: vT is h-major
      acc[h4] = WMMA_F16(aw, bv, acc[h4]);
    }
    wave_lds_fence_war();   // keep next iteration's stores after these reads
  }

#pragma unroll
  for (int h4 = 0; h4 < 4; ++h4)
#pragma unroll
    for (int r = 0; r < 8; ++r)
      out[(size_t)(b * TT + t0 + r + lhalf * 8) * HH + h4 * 16 + lmod] = acc[h4][r];
}

// ---------------------------------------------------------------------------
extern "C" void kernel_launch(void* const* d_in, const int* in_sizes, int n_in,
                              void* d_out, int out_size, void* d_ws, size_t ws_size,
                              hipStream_t stream) {
  (void)in_sizes; (void)n_in; (void)out_size; (void)ws_size;

  const float* x  = (const float*)d_in[0];
  const float* Wq = (const float*)d_in[1];
  const float* Wk = (const float*)d_in[2];
  // d_in[3] (Wv) intentionally unused: reference computes v = x @ Wk.
  float* out = (float*)d_out;

  // Workspace layout (~12.2 MB total):
  //   qh: B*T*H f16 (4 MB) | kh: 4 MB | vT: 4 MB | Z: B*T f32 (128 KB)
  _Float16* qh = (_Float16*)d_ws;
  _Float16* kh = qh + (size_t)BB * TT * HH;
  _Float16* vT = kh + (size_t)BB * TT * HH;
  float*    Z  = (float*)(vT + (size_t)BB * TT * HH);

  proj_kernel<<<(BB * TT) / 256, 256, 0, stream>>>(x, Wq, Wk, qh, kh, vT);
  zsum_kernel<<<(BB * (TT / 16)) / 8, 256, 0, stream>>>(qh, kh, Z);
  attn_out_kernel<<<(BB * (TT / 16)) / 8, 256, 0, stream>>>(qh, kh, vT, Z, out);
}
